// ROIAlign_9552007266619
// MI455X (gfx1250) — matrix operand order repaired and checked by reading, not compiled, not run on hardware
//
#include <hip/hip_runtime.h>

// ROI Align for MI455X (gfx1250, wave32).
//
// Roofline: ~0.4 GFLOP vs ~71 MB HBM traffic (input 20.5MB is L2-resident,
// output 51MB) -> memory bound, ~3us floor at 23.3 TB/s. Strategy:
//  - channel-contiguous (NHWC) gathers: a wave reads 16 consecutive channels
//    per pixel -> coalesced bursts, pixel reuse captured by WGP$/L2.
//  - y-interp (2 bilinear rows per pooled row, 4 taps) in VALU, weights
//    pre-scaled by the 0.5 y-pooling factor.
//  - x-interp + x-pooling fused into ONE v_wmma_f32_16x16x32_f16 per
//    (output row, 16-channel tile): K=32 covers the whole x-window
//    (<= 21 cols for bw <= 0.4), M = ow (7 of 16 rows used), N = channels.
//    f16 in / f32 accumulate keeps error ~1e-3 on N(0,1) data.
//
// Lane mapping uses a contiguous K order, not AMD's table order, for B (32x16 f16): lanes 0-15
// hold K=0-15 at N=lane, lanes 16-31 hold K=16-31 at N=lane-16. So lane L
// owns channel tile_base+(L&15) and computes U only for its K half; the B
// operand forms directly in registers, no LDS shuffle required.

typedef __attribute__((ext_vector_type(16))) _Float16 v16h;
typedef __attribute__((ext_vector_type(8)))  float    v8f;

#define H_   50
#define W_   50
#define C_   256
#define OHW  7     // output 7x7
#define CH_  14    // sampling grid 14x14 (2x2 pooled)

union V16H { _Float16 h[16]; v16h v; };

__global__ __launch_bounds__(512)
void roi_align_wmma_kernel(const float* __restrict__ img,     // [8,50,50,256]
                           const float* __restrict__ boxes,   // [N,4] (cy,cx,bh,bw)
                           const int*   __restrict__ box_idx, // [N]
                           float*       __restrict__ out)     // [N,7,7,256]
{
    const int n    = blockIdx.x;
    const int tid  = threadIdx.x;
    const int wv   = tid >> 5;        // wave id 0..15 -> channel tile
    const int lane = tid & 31;
    const int m    = lane & 15;       // WMMA N index (channel offset) and M row (=ow)
    const int hi   = lane >> 4;       // K half: 0 -> K 0..15, 1 -> K 16..31
    const int c    = wv * 16 + m;     // this lane's channel

    // ---- box geometry (wave-uniform) ----
    const float cy = boxes[n*4+0], cx = boxes[n*4+1];
    const float bh = boxes[n*4+2], bw = boxes[n*4+3];
    const int   b  = box_idx[n];

    const float y1 = cy - 0.5f*bh, x1 = cx - 0.5f*bw;
    const float y2 = cy + 0.5f*bh, x2 = cx + 0.5f*bw;
    const float bin_h = (y2 - y1) * (1.0f/OHW);
    const float bin_w = (x2 - x1) * (1.0f/OHW);
    const float gy1 = y1 + 0.25f*bin_h, gy2 = y2 - 0.25f*bin_h;
    const float gx1 = x1 + 0.25f*bin_w, gx2 = x2 - 0.25f*bin_w;
    const float oy = gy1 * (float)(H_-1);
    const float ox = gx1 * (float)(W_-1);
    const float sy = (gy2 - gy1) * (float)(H_-1) / (float)(CH_-1);
    const float sx = (gx2 - gx1) * (float)(W_-1) / (float)(CH_-1);

    // x-window start: xs(i) is increasing, so min tap column is at ix=0
    const int x0w = min(max((int)floorf(ox), 0), W_-1);

    // ---- A matrix: pooled x-weights, 16x32 f16 (per-box constant) ----
    // lane L<16:  slots 0..15 = K 0..15  (row M = m)
    // lane L>=16: slots 0..15 = K 16..31  (row M = m)
    float aw[16];
    #pragma unroll
    for (int e = 0; e < 16; ++e) aw[e] = 0.0f;
    if (m < OHW) {
        #pragma unroll
        for (int t = 0; t < 2; ++t) {
            const int   ix = 2*m + t;
            const float xs = ox + (float)ix * sx;
            const float xf = floorf(xs);
            const float dx = xs - xf;
            const int   pl = min(max((int)xf,     0), W_-1);
            const int   ph = min(max((int)xf + 1, 0), W_-1);
            const int   kl = pl - x0w;
            const int   kh = ph - x0w;
            #pragma unroll
            for (int e = 0; e < 16; ++e) {
                const int k = hi * 16 + e;
                if (k == kl) aw[e] += 0.5f * (1.0f - dx);
                if (k == kh) aw[e] += 0.5f * dx;       // pl==ph merges taps correctly
            }
        }
    }
    V16H A;
    #pragma unroll
    for (int e = 0; e < 16; ++e) A.h[e] = (_Float16)aw[e];

    const int imgbase = b * (H_*W_*C_);
    const int xbase   = x0w + hi*16;   // this lane's K-half window origin

    for (int oh = 0; oh < OHW; ++oh) {
        // y taps for the two sample rows pooled into this output row;
        // 0.5 pooling factor folded into the weights.
        float wy[4]; int rr[4];
        #pragma unroll
        for (int t = 0; t < 2; ++t) {
            const float ys = oy + (float)(2*oh + t) * sy;
            const float yf = floorf(ys);
            const float dy = ys - yf;
            rr[2*t]   = min(max((int)yf,     0), H_-1);
            rr[2*t+1] = min(max((int)yf + 1, 0), H_-1);
            wy[2*t]   = 0.5f * (1.0f - dy);
            wy[2*t+1] = 0.5f * dy;
        }
        const float* p0 = img + imgbase + rr[0]*(W_*C_) + c;
        const float* p1 = img + imgbase + rr[1]*(W_*C_) + c;
        const float* p2 = img + imgbase + rr[2]*(W_*C_) + c;
        const float* p3 = img + imgbase + rr[3]*(W_*C_) + c;

        // B matrix: U[x, c] for this lane's 16 K positions (y-lerped rows)
        V16H B;
        #pragma unroll
        for (int e = 0; e < 16; ++e) {
            const int col = min(xbase + e, W_-1);   // A weight is 0 past window
            const int off = col * C_;
            const float u = wy[0]*p0[off] + wy[1]*p1[off]
                          + wy[2]*p2[off] + wy[3]*p3[off];
            B.h[e] = (_Float16)u;
        }

        // out[oh, ow=0..6, c-tile] = A (16x32) x B (32x16), f32 accumulate
        v8f cz = {};
        v8f d = __builtin_amdgcn_wmma_f32_16x16x32_f16(
                    /*neg_a=*/false, A.v, /*neg_b=*/false, B.v,
                    /*c_mod=*/(short)0, cz, /*reuse_a=*/false, /*reuse_b=*/false);

        // D layout: rows M=0..6 live in VGPRs 0..6 of lanes 0..15 (N = m)
        if (hi == 0) {
            float* o = out + ((n*OHW + oh)*OHW)*C_ + c;
            #pragma unroll
            for (int ow = 0; ow < OHW; ++ow) o[ow*C_] = d[ow];
        }
    }
}

extern "C" void kernel_launch(void* const* d_in, const int* in_sizes, int n_in,
                              void* d_out, int out_size, void* d_ws, size_t ws_size,
                              hipStream_t stream) {
    const float* img     = (const float*)d_in[0];
    const float* boxes   = (const float*)d_in[1];
    const int*   box_idx = (const int*)d_in[2];
    float*       out     = (float*)d_out;

    const int n_boxes = in_sizes[2];   // box_indices count (N = 1000)
    (void)n_in; (void)out_size; (void)d_ws; (void)ws_size;

    roi_align_wmma_kernel<<<n_boxes, 512, 0, stream>>>(img, boxes, box_idx, out);
}
